// ProposalModule_64742337020315
// MI455X (gfx1250) — matrix operand order repaired and checked by reading, not compiled, not run on hardware
//
#include <hip/hip_runtime.h>

// ---------------- constants ----------------
#define Bn      32
#define Nn      2048
#define CSEED   256
#define NPOINTc 256
#define NSAMPLEc 32
#define RADIUSc 12.0f
#define NHc     12
#define MT      (Bn * NPOINTc * NSAMPLEc)   // 262144 columns for MLP GEMMs
#define MC      (Bn * NPOINTc)              // 8192 columns for conv GEMMs
#define K0P     288                         // 259 padded to 9*32

typedef _Float16 half_t;
typedef __attribute__((ext_vector_type(16))) _Float16 v16h;
typedef __attribute__((ext_vector_type(8)))  _Float16 v8h;
typedef __attribute__((ext_vector_type(8)))  float    v8f;
typedef __attribute__((ext_vector_type(4)))  int      v4i;
typedef __attribute__((address_space(1))) v4i* g_v4i_p;
typedef __attribute__((address_space(3))) v4i* l_v4i_p;

#if defined(__has_builtin)
#if __has_builtin(__builtin_amdgcn_global_load_async_to_lds_b128)
#define USE_ASYNC_LDS 1
#endif
#endif

// X kept in global memory pre-swizzled into WMMA B-fragment tile order:
//   tile(kt, mt) = 4096 halves laid out [sub(8)][lane(32)][j(16)]
__device__ __forceinline__ size_t xsw_off(int k, int col, int Mtot) {
  int kt = k >> 5, kk = k & 31;
  int mt = col >> 7, mm = col & 127;
  int hs = kk >> 4, j = kk & 15;
  return ((size_t)kt * (size_t)(Mtot >> 7) + (size_t)mt) * 4096 +
         (size_t)((mm >> 4) * 512 + (mm & 15) * 16 + hs * 256 + j);
}

// Y kept in D-fragment order: off(ch,col) = (((mt*8+s)*8 + w)*32 + n+16*hi)*8 + r
// with ch = 16*w + 8*hi + r,  col = mt*128 + s*16 + n.
__device__ __forceinline__ size_t yfrag_off(int ch, int col) {
  int w = ch >> 4, hi = (ch >> 3) & 1, r = ch & 7;
  int mt = col >> 7, s = (col >> 4) & 7, n = col & 15;
  return ((((size_t)(mt * 8 + s) * 8 + w) * 32) + (n + 16 * hi)) * 8 + r;
}

// ---------------- FPS: one block per batch ----------------
__global__ void fps_kernel(const float* __restrict__ xyz,
                           int* __restrict__ fps_inds,
                           float* __restrict__ new_xyz) {
  int b = blockIdx.x;
  __shared__ float px[Nn], py[Nn], pz[Nn], dist[Nn];
  __shared__ float rbest[256];
  __shared__ int   ribx[256];
  const float* base = xyz + (size_t)b * Nn * 3;
  for (int n = threadIdx.x; n < Nn; n += 256) {
    px[n] = base[n * 3 + 0];
    py[n] = base[n * 3 + 1];
    pz[n] = base[n * 3 + 2];
    dist[n] = 1e10f;
  }
  __syncthreads();
  int last = 0;
  if (threadIdx.x == 0) {
    fps_inds[b * NPOINTc] = 0;
    new_xyz[(size_t)(b * NPOINTc) * 3 + 0] = px[0];
    new_xyz[(size_t)(b * NPOINTc) * 3 + 1] = py[0];
    new_xyz[(size_t)(b * NPOINTc) * 3 + 2] = pz[0];
  }
  for (int it = 1; it < NPOINTc; it++) {
    float lx = px[last], ly = py[last], lz = pz[last];
    float bm = -1.0f; int bi = 0;
    for (int n = threadIdx.x; n < Nn; n += 256) {
      float dx = px[n] - lx, dy = py[n] - ly, dz = pz[n] - lz;
      float d = dx * dx + dy * dy + dz * dz;
      float dn = fminf(dist[n], d);
      dist[n] = dn;
      if (dn > bm) { bm = dn; bi = n; }
    }
    rbest[threadIdx.x] = bm; ribx[threadIdx.x] = bi;
    __syncthreads();
    for (int st = 128; st > 0; st >>= 1) {
      if (threadIdx.x < st) {
        float o = rbest[threadIdx.x + st]; int oi = ribx[threadIdx.x + st];
        if (o > rbest[threadIdx.x] ||
            (o == rbest[threadIdx.x] && oi < ribx[threadIdx.x])) {
          rbest[threadIdx.x] = o; ribx[threadIdx.x] = oi;
        }
      }
      __syncthreads();
    }
    last = ribx[0];
    if (threadIdx.x == 0) {
      fps_inds[b * NPOINTc + it] = last;
      new_xyz[(size_t)(b * NPOINTc + it) * 3 + 0] = px[last];
      new_xyz[(size_t)(b * NPOINTc + it) * 3 + 1] = py[last];
      new_xyz[(size_t)(b * NPOINTc + it) * 3 + 2] = pz[last];
    }
    __syncthreads();
  }
}

// ---------------- ball query ----------------
__global__ void ballquery_kernel(const float* __restrict__ xyz,
                                 const float* __restrict__ new_xyz,
                                 int* __restrict__ idx) {
  int i = blockIdx.x * blockDim.x + threadIdx.x;
  if (i >= MC) return;
  int b = i >> 8;
  float qx = new_xyz[(size_t)i * 3 + 0];
  float qy = new_xyz[(size_t)i * 3 + 1];
  float qz = new_xyz[(size_t)i * 3 + 2];
  const float* base = xyz + (size_t)b * Nn * 3;
  int* op = idx + (size_t)i * NSAMPLEc;
  int cnt = 0;
  const float r2 = RADIUSc * RADIUSc;
  for (int n = 0; n < Nn && cnt < NSAMPLEc; n++) {
    float dx = base[n * 3 + 0] - qx;
    float dy = base[n * 3 + 1] - qy;
    float dz = base[n * 3 + 2] - qz;
    if (dx * dx + dy * dy + dz * dz < r2) op[cnt++] = n;
  }
  int first = (cnt > 0) ? op[0] : (Nn - 1);
  for (; cnt < NSAMPLEc; cnt++) op[cnt] = first;
}

// ---------------- gather + f16 into swizzled X0 ----------------
__global__ void gather_kernel(const float* __restrict__ xyz,
                              const float* __restrict__ features,
                              const float* __restrict__ new_xyz,
                              const int* __restrict__ idx,
                              half_t* __restrict__ X0) {
  int t = blockIdx.x * blockDim.x + threadIdx.x;
  if (t >= MT) return;
  int b = t >> 13;
  int m = t & 8191;
  int p = m >> 5;
  int s = m & 31;
  int pi = idx[((size_t)b * NPOINTc + p) * NSAMPLEc + s];
  const float* pw = xyz + ((size_t)b * Nn + pi) * 3;
  const float* q  = new_xyz + ((size_t)b * NPOINTc + p) * 3;
  const float inv = 1.0f / RADIUSc;
  float dx = (pw[0] - q[0]) * inv;
  float dy = (pw[1] - q[1]) * inv;
  float dz = (pw[2] - q[2]) * inv;
  const float* fb = features + (size_t)b * CSEED * Nn + pi;
  int mt = t >> 7, mm = t & 127;
  size_t colbase = (size_t)((mm >> 4) * 512 + (mm & 15) * 16);
  const int mtiles = MT >> 7;
  for (int g = 0; g < K0P / 16; g++) {
    v16h v;
    #pragma unroll
    for (int e = 0; e < 16; e++) {
      int k = g * 16 + e;
      float val;
      if (k >= 3) val = (k < 259) ? fb[(size_t)(k - 3) * Nn] : 0.0f;
      else        val = (k == 0) ? dx : (k == 1) ? dy : dz;
      v[e] = (half_t)val;
    }
    size_t addr = ((size_t)(g >> 1) * mtiles + mt) * 4096 + colbase +
                  (size_t)(g & 1) * 256;
    *(v16h*)(X0 + addr) = v;
  }
}

// ---------------- weight prep ----------------
__global__ void prep_weights(const float* __restrict__ w0, const float* __restrict__ w1,
                             const float* __restrict__ w2, const float* __restrict__ wc1,
                             const float* __restrict__ wc2, const float* __restrict__ w3,
                             half_t* __restrict__ W0h, half_t* __restrict__ W1h,
                             half_t* __restrict__ W2h, half_t* __restrict__ Wc1h,
                             half_t* __restrict__ Wc2h, half_t* __restrict__ W3h) {
  int i = blockIdx.x * blockDim.x + threadIdx.x;
  if (i < 128 * K0P) {
    int o = i / K0P, k = i % K0P;
    W0h[i] = (half_t)(k < 259 ? w0[(size_t)o * 259 + k] : 0.0f);
  }
  if (i < 128 * 128) {
    W1h[i]  = (half_t)w1[i];
    W2h[i]  = (half_t)w2[i];
    Wc1h[i] = (half_t)wc1[i];
    Wc2h[i] = (half_t)wc2[i];
    int o = i >> 7;
    W3h[i] = (half_t)(o < 77 ? w3[i] : 0.0f);
  }
}

// ---------------- WMMA GEMM: D-fragment output ----------------
template <bool F32OUT>
__global__ void gemm_wmma(const half_t* __restrict__ W,    // [128][K] linear
                          const half_t* __restrict__ Xsw,  // swizzled tiles
                          void* __restrict__ Yout,         // fragment layout
                          int K, int Mtot) {
  __shared__ __attribute__((aligned(32))) half_t lA[2][4096];
  __shared__ __attribute__((aligned(32))) half_t lB[2][4096];
  const int t    = threadIdx.x;
  const int wave = t >> 5;
  const int lane = t & 31;
  const int mt   = blockIdx.x;
  const int mtiles = Mtot >> 7;
  const int KT   = K >> 5;

  v8f acc[8];
  #pragma unroll
  for (int s = 0; s < 8; s++)
    #pragma unroll
    for (int r = 0; r < 8; r++) acc[s][r] = 0.0f;

  const int so = t >> 1, shs = t & 1;
  const half_t* wrow = W + (size_t)so * K + 16 * shs;
  const int aoff = (so >> 4) * 512 + ((so & 15) + 16 * shs) * 16;

#ifdef USE_ASYNC_LDS
  auto stage = [&](int kt, int buf) {
    const half_t* bs = Xsw + ((size_t)kt * mtiles + mt) * 4096 + t * 16;
    const half_t* as = wrow + kt * 32;
    __builtin_amdgcn_global_load_async_to_lds_b128(
        (g_v4i_p)bs, (l_v4i_p)&lB[buf][t * 16], 0, 0);
    __builtin_amdgcn_global_load_async_to_lds_b128(
        (g_v4i_p)(bs + 8), (l_v4i_p)&lB[buf][t * 16 + 8], 0, 0);
    __builtin_amdgcn_global_load_async_to_lds_b128(
        (g_v4i_p)as, (l_v4i_p)&lA[buf][aoff], 0, 0);
    __builtin_amdgcn_global_load_async_to_lds_b128(
        (g_v4i_p)(as + 8), (l_v4i_p)&lA[buf][aoff + 8], 0, 0);
  };
  auto wait_async = [&]() {
#if __has_builtin(__builtin_amdgcn_s_wait_asynccnt)
    __builtin_amdgcn_s_wait_asynccnt(0);
#else
    asm volatile("s_wait_asynccnt 0x0" ::: "memory");
#endif
  };
  stage(0, 0);
  wait_async();
#else
  v8h ra0, ra1; v16h rb;
  auto gload = [&](int kt) {
    ra0 = *(const v8h*)(wrow + kt * 32);
    ra1 = *(const v8h*)(wrow + kt * 32 + 8);
    rb  = *(const v16h*)(Xsw + ((size_t)kt * mtiles + mt) * 4096 + t * 16);
  };
  auto lstore = [&](int buf) {
    *(v8h*)&lA[buf][aoff]     = ra0;
    *(v8h*)&lA[buf][aoff + 8] = ra1;
    *(v16h*)&lB[buf][t * 16]  = rb;
  };
  gload(0);
  lstore(0);
#endif
  __syncthreads();

  for (int kt = 0; kt < KT; kt++) {
    const int buf = kt & 1;
    if (kt + 1 < KT) {
#ifdef USE_ASYNC_LDS
      stage(kt + 1, buf ^ 1);
#else
      gload(kt + 1);
#endif
      if (kt + 2 < KT)
        __builtin_prefetch(Xsw + ((size_t)(kt + 2) * mtiles + mt) * 4096 + t * 16, 0, 3);
    }
    v16h a  = *(const v16h*)&lA[buf][wave * 512 + lane * 16];
    v16h b0 = *(const v16h*)&lB[buf][0 * 512 + lane * 16];
    v16h b1 = *(const v16h*)&lB[buf][1 * 512 + lane * 16];
    v16h b2 = *(const v16h*)&lB[buf][2 * 512 + lane * 16];
    v16h b3 = *(const v16h*)&lB[buf][3 * 512 + lane * 16];
    v16h b4 = *(const v16h*)&lB[buf][4 * 512 + lane * 16];
    v16h b5 = *(const v16h*)&lB[buf][5 * 512 + lane * 16];
    v16h b6 = *(const v16h*)&lB[buf][6 * 512 + lane * 16];
    v16h b7 = *(const v16h*)&lB[buf][7 * 512 + lane * 16];
    acc[0] = __builtin_amdgcn_wmma_f32_16x16x32_f16(false, a, false, b0, (short)0, acc[0], false, false);
    acc[1] = __builtin_amdgcn_wmma_f32_16x16x32_f16(false, a, false, b1, (short)0, acc[1], false, false);
    acc[2] = __builtin_amdgcn_wmma_f32_16x16x32_f16(false, a, false, b2, (short)0, acc[2], false, false);
    acc[3] = __builtin_amdgcn_wmma_f32_16x16x32_f16(false, a, false, b3, (short)0, acc[3], false, false);
    acc[4] = __builtin_amdgcn_wmma_f32_16x16x32_f16(false, a, false, b4, (short)0, acc[4], false, false);
    acc[5] = __builtin_amdgcn_wmma_f32_16x16x32_f16(false, a, false, b5, (short)0, acc[5], false, false);
    acc[6] = __builtin_amdgcn_wmma_f32_16x16x32_f16(false, a, false, b6, (short)0, acc[6], false, false);
    acc[7] = __builtin_amdgcn_wmma_f32_16x16x32_f16(false, a, false, b7, (short)0, acc[7], false, false);
#ifdef USE_ASYNC_LDS
    if (kt + 1 < KT) wait_async();
#else
    if (kt + 1 < KT) lstore(buf ^ 1);
#endif
    __syncthreads();
  }

  // ---- store D in fragment order: one vector store per subtile per lane ----
  #pragma unroll
  for (int s = 0; s < 8; s++) {
    size_t base = ((((size_t)(mt * 8 + s) * 8 + wave) * 32) + lane) * 8;
    if (F32OUT) {
      *(v8f*)((float*)Yout + base) = acc[s];
    } else {
      v8h h;
      #pragma unroll
      for (int r = 0; r < 8; r++) h[r] = (half_t)acc[s][r];
      *(v8h*)((half_t*)Yout + base) = h;
    }
  }
}

// ---------------- BN stats phase 1: per-block partial sums (deterministic) ------
__global__ void bn_stats_partial(const half_t* __restrict__ Yf,
                                 double* __restrict__ part, int Mtot) {
  size_t base = (size_t)blockIdx.x * (size_t)Mtot;   // 128 blocks cover 128*Mtot
  int t = threadIdx.x;
  double s0[8], s1[8];
  #pragma unroll
  for (int r = 0; r < 8; r++) { s0[r] = 0.0; s1[r] = 0.0; }
  for (size_t e = (size_t)t * 8; e < (size_t)Mtot; e += 2048) {
    v8h y = *(const v8h*)(Yf + base + e);
    #pragma unroll
    for (int r = 0; r < 8; r++) {
      double v = (double)(float)y[r];
      s0[r] += v; s1[r] += v * v;
    }
  }
  __shared__ double sh0[256][8], sh1[256][8];
  #pragma unroll
  for (int r = 0; r < 8; r++) { sh0[t][r] = s0[r]; sh1[t][r] = s1[r]; }
  __syncthreads();
  if (t < 128) {   // t == channel
    int w = t >> 4, hi = (t >> 3) & 1, r = t & 7;
    int tb = w * 32 + hi * 16;
    double a0 = 0.0, a1 = 0.0;
    for (int i = 0; i < 16; i++) { a0 += sh0[tb + i][r]; a1 += sh1[tb + i][r]; }
    part[(size_t)blockIdx.x * 256 + t * 2 + 0] = a0;
    part[(size_t)blockIdx.x * 256 + t * 2 + 1] = a1;
  }
}

// ---------------- BN stats phase 2: fold gamma/beta -> stats3 ----------------
__global__ void bn_stats_final(const double* __restrict__ part,
                               const float* __restrict__ g,
                               const float* __restrict__ be,
                               float* __restrict__ stats3, int Mtot) {
  int ch = threadIdx.x;   // 128 threads, 1 block
  double a0 = 0.0, a1 = 0.0;
  for (int b = 0; b < 128; b++) {
    a0 += part[(size_t)b * 256 + ch * 2 + 0];
    a1 += part[(size_t)b * 256 + ch * 2 + 1];
  }
  double mean = a0 / (double)Mtot;
  double var  = a1 / (double)Mtot - mean * mean;
  float rstd = rsqrtf((float)var + 1e-5f);
  stats3[ch * 4 + 0] = (float)mean;
  stats3[ch * 4 + 1] = rstd * g[ch];
  stats3[ch * 4 + 2] = be[ch];
}

// ---------------- BN apply + ReLU: fragment in, swizzled out, vectorized ------
__global__ void bn_apply(const half_t* __restrict__ Yf,
                         const float* __restrict__ stats3,
                         half_t* __restrict__ Xo, int Mtot) {
  size_t i8 = ((size_t)blockIdx.x * blockDim.x + threadIdx.x) * 8;
  if (i8 >= (size_t)128 * (size_t)Mtot) return;
  size_t qq = i8 >> 3;
  int l  = (int)(qq & 31); int hi = l >> 4; int n = l & 15;
  size_t q2 = qq >> 5;
  int w  = (int)(q2 & 7);
  size_t q3 = q2 >> 3;
  int s  = (int)(q3 & 7);
  int mt = (int)(q3 >> 3);
  int col = mt * 128 + s * 16 + n;
  int ch0 = 16 * w + 8 * hi;
  v8h y = *(const v8h*)(Yf + i8);
  v8h o;
  #pragma unroll
  for (int r = 0; r < 8; r++) {
    int ch = ch0 + r;
    float vv = ((float)y[r] - stats3[ch * 4]) * stats3[ch * 4 + 1] + stats3[ch * 4 + 2];
    o[r] = (half_t)fmaxf(vv, 0.0f);
  }
  *(v8h*)(Xo + xsw_off(ch0, col, Mtot)) = o;
}

// ---------------- max pool over S=32 (swizzled in/out) ----------------
__global__ void maxpool_kernel(const half_t* __restrict__ X, half_t* __restrict__ fG) {
  int i = blockIdx.x * blockDim.x + threadIdx.x;
  if (i >= 128 * MC) return;
  int ch = i >> 13;
  int bp = i & 8191;
  float m = -1e30f;
  #pragma unroll
  for (int s = 0; s < NSAMPLEc; s++)
    m = fmaxf(m, (float)X[xsw_off(ch, bp * NSAMPLEc + s, MT)]);
  fG[xsw_off(ch, bp, MC)] = (half_t)m;
}

// ---------------- final assembly (Y3 fragment-ordered f32) ----------------
__global__ void assemble_kernel(const float* __restrict__ Y3, const float* __restrict__ b3,
                                const float* __restrict__ new_xyz, float* __restrict__ out) {
  int i = blockIdx.x * blockDim.x + threadIdx.x;
  if (i >= MC) return;
  float t[77];
  #pragma unroll
  for (int c = 0; c < 77; c++) t[c] = Y3[yfrag_off(c, i)] + b3[c];
  float* o = out + (size_t)i * 113;
  o[0] = t[0]; o[1] = t[1];
  o[2] = new_xyz[(size_t)i * 3 + 0] + t[2];
  o[3] = new_xyz[(size_t)i * 3 + 1] + t[3];
  o[4] = new_xyz[(size_t)i * 3 + 2] + t[4];
  #pragma unroll
  for (int c = 0; c < 36; c++) o[5 + c]  = t[5 + c];
  #pragma unroll
  for (int c = 0; c < 36; c++) o[41 + c] = t[41 + c];
  const float f = 3.14159265358979323846f / (float)NHc;
  #pragma unroll
  for (int c = 0; c < 36; c++) o[77 + c] = t[41 + c] * f;
}

// ---------------- host launcher ----------------
extern "C" void kernel_launch(void* const* d_in, const int* in_sizes, int n_in,
                              void* d_out, int out_size, void* d_ws, size_t ws_size,
                              hipStream_t stream) {
  const float* xyz      = (const float*)d_in[0];
  const float* features = (const float*)d_in[1];
  const float* mlp_w0 = (const float*)d_in[2];
  const float* mlp_g0 = (const float*)d_in[3];
  const float* mlp_b0 = (const float*)d_in[4];
  const float* mlp_w1 = (const float*)d_in[5];
  const float* mlp_g1 = (const float*)d_in[6];
  const float* mlp_b1 = (const float*)d_in[7];
  const float* mlp_w2 = (const float*)d_in[8];
  const float* mlp_g2 = (const float*)d_in[9];
  const float* mlp_b2 = (const float*)d_in[10];
  const float* conv1_w = (const float*)d_in[11];
  const float* bn1_g   = (const float*)d_in[13];
  const float* bn1_b   = (const float*)d_in[14];
  const float* conv2_w = (const float*)d_in[15];
  const float* bn2_g   = (const float*)d_in[17];
  const float* bn2_b   = (const float*)d_in[18];
  const float* conv3_w = (const float*)d_in[19];
  const float* conv3_b = (const float*)d_in[20];
  float* out = (float*)d_out;

  char* ws = (char*)d_ws;
  size_t off = 0;
  auto carve = [&](size_t bytes) -> char* {
    char* p = ws + off;
    off = (off + bytes + 255) & ~(size_t)255;
    return p;
  };
  int*    fps_inds = (int*)   carve((size_t)Bn * NPOINTc * 4);
  float*  new_xyz  = (float*) carve((size_t)Bn * NPOINTc * 3 * 4);
  int*    idx      = (int*)   carve((size_t)Bn * NPOINTc * NSAMPLEc * 4);
  half_t* W0h  = (half_t*) carve((size_t)128 * K0P * 2);
  half_t* W1h  = (half_t*) carve((size_t)128 * 128 * 2);
  half_t* W2h  = (half_t*) carve((size_t)128 * 128 * 2);
  half_t* Wc1h = (half_t*) carve((size_t)128 * 128 * 2);
  half_t* Wc2h = (half_t*) carve((size_t)128 * 128 * 2);
  half_t* W3h  = (half_t*) carve((size_t)128 * 128 * 2);
  half_t* X0   = (half_t*) carve((size_t)K0P * MT * 2);   // 151 MB (swizzled)
  half_t* XB   = (half_t*) carve((size_t)128 * MT * 2);   // 67 MB  (swizzled)
  half_t* Yh   = (half_t*) carve((size_t)128 * MT * 2);   // 67 MB  (fragment)
  double* part = (double*) carve((size_t)128 * 256 * 8);  // stats partials
  float*  st3  = (float*)  carve((size_t)128 * 4 * 4);
  half_t* fG   = (half_t*) carve((size_t)128 * MC * 2);   // swizzled
  half_t* g1b  = (half_t*) carve((size_t)128 * MC * 2);   // swizzled
  half_t* Yc   = (half_t*) carve((size_t)128 * MC * 2);   // fragment
  float*  Y3   = (float*)  carve((size_t)128 * MC * 4);   // fragment
  (void)ws_size; (void)in_sizes; (void)n_in; (void)out_size; (void)fps_inds;

  prep_weights<<<(128 * K0P + 255) / 256, 256, 0, stream>>>(
      mlp_w0, mlp_w1, mlp_w2, conv1_w, conv2_w, conv3_w,
      W0h, W1h, W2h, Wc1h, Wc2h, W3h);
  fps_kernel<<<Bn, 256, 0, stream>>>(xyz, fps_inds, new_xyz);
  ballquery_kernel<<<MC / 256, 256, 0, stream>>>(xyz, new_xyz, idx);
  gather_kernel<<<MT / 256, 256, 0, stream>>>(xyz, features, new_xyz, idx, X0);

  // MLP layer 0
  gemm_wmma<false><<<MT / 128, 256, 0, stream>>>(W0h, X0, Yh, K0P, MT);
  bn_stats_partial<<<128, 256, 0, stream>>>(Yh, part, MT);
  bn_stats_final<<<1, 128, 0, stream>>>(part, mlp_g0, mlp_b0, st3, MT);
  bn_apply<<<(128 * MT / 8) / 256, 256, 0, stream>>>(Yh, st3, XB, MT);
  // MLP layer 1
  gemm_wmma<false><<<MT / 128, 256, 0, stream>>>(W1h, XB, Yh, 128, MT);
  bn_stats_partial<<<128, 256, 0, stream>>>(Yh, part, MT);
  bn_stats_final<<<1, 128, 0, stream>>>(part, mlp_g1, mlp_b1, st3, MT);
  bn_apply<<<(128 * MT / 8) / 256, 256, 0, stream>>>(Yh, st3, X0, MT);
  // MLP layer 2
  gemm_wmma<false><<<MT / 128, 256, 0, stream>>>(W2h, X0, Yh, 128, MT);
  bn_stats_partial<<<128, 256, 0, stream>>>(Yh, part, MT);
  bn_stats_final<<<1, 128, 0, stream>>>(part, mlp_g2, mlp_b2, st3, MT);
  bn_apply<<<(128 * MT / 8) / 256, 256, 0, stream>>>(Yh, st3, XB, MT);
  // max pool
  maxpool_kernel<<<(128 * MC) / 256, 256, 0, stream>>>(XB, fG);

  // conv1 (bias cancels under BN)
  gemm_wmma<false><<<MC / 128, 256, 0, stream>>>(Wc1h, fG, Yc, 128, MC);
  bn_stats_partial<<<128, 256, 0, stream>>>(Yc, part, MC);
  bn_stats_final<<<1, 128, 0, stream>>>(part, bn1_g, bn1_b, st3, MC);
  bn_apply<<<(128 * MC / 8) / 256, 256, 0, stream>>>(Yc, st3, g1b, MC);
  // conv2
  gemm_wmma<false><<<MC / 128, 256, 0, stream>>>(Wc2h, g1b, Yc, 128, MC);
  bn_stats_partial<<<128, 256, 0, stream>>>(Yc, part, MC);
  bn_stats_final<<<1, 128, 0, stream>>>(part, bn2_g, bn2_b, st3, MC);
  bn_apply<<<(128 * MC / 8) / 256, 256, 0, stream>>>(Yc, st3, fG, MC);
  // conv3 (padded to 128 rows, f32 fragment output)
  gemm_wmma<true><<<MC / 128, 256, 0, stream>>>(W3h, fG, (void*)Y3, 128, MC);
  assemble_kernel<<<MC / 256, 256, 0, stream>>>(Y3, conv3_b, new_xyz, out);
}
